// GRUcell_4329327034834
// MI455X (gfx1250) — matrix operand order repaired and checked by reading, not compiled, not run on hardware
//
#include <hip/hip_runtime.h>

// ---------------------------------------------------------------------------
// Fused GRU for MI455X (gfx1250, wave32).
// grid = B/16 workgroups x 256 threads (8 waves). Wave w owns gate columns
// [16w,16w+16). Weights live in LDS as f16 (~238KB of the 320KB WGP LDS),
// matmuls are v_wmma_f32_16x16x32_f16, recurrence state H carried in f32
// registers (C-layout ownership is static per lane), f16 mirror in LDS for
// the A operand. X tile for t+1 is prefetched into registers during step t.
// 3 barriers / step; 28 WMMAs / wave / step in chains of depth <= 4.
// ---------------------------------------------------------------------------

#define HS        128
#define BM        16            // batch rows per workgroup (= WMMA M)
#define NW        8             // waves per workgroup (= 128/16 gate col tiles)
#define NTHREADS  (NW * 32)
#define KSTEPS    4             // 128 / 32 (WMMA K)
#define SRH       (HS + 8)      // f16 row stride (elements) -> 272B rows

typedef __attribute__((ext_vector_type(16))) _Float16 v16h;
typedef __attribute__((ext_vector_type(8)))  _Float16 v8h;
typedef __attribute__((ext_vector_type(8)))  float    v8f;
typedef __attribute__((ext_vector_type(4)))  float    v4f;

union F16x16 { v16h v; v8h h[2]; };

// ---- LDS layout (bytes) ----------------------------------------------------
#define OFF_W    0
#define SZ_W     (7 * HS * SRH * 2)           // 7 f16 weight matrices: 243712B
#define OFF_X    (OFF_W + SZ_W)
#define SZ_T16   (BM * SRH * 2)               // 4352B
#define OFF_H16  (OFF_X  + SZ_T16)
#define OFF_RH   (OFF_H16 + SZ_T16)
#define SMEM_BYTES (OFF_RH + SZ_T16)          // 256768B (< 320KB WGP LDS)

struct GruParams {
  const float* X;  const float* H0;
  const float* Wxz; const float* bxz; const float* Whz; const float* bhz;
  const float* Wxr; const float* bxr; const float* Whr; const float* bhr;
  const float* Wxh; const float* bxh; const float* Whh; const float* bhh;
  const float* Whq; const float* bhq;
  float* Y; int Bsz; int T;
};

__device__ __forceinline__ float sigm(float x) { return 1.0f / (1.0f + __expf(-x)); }
__device__ __forceinline__ float tanh_fast(float x) {
  float e = __expf(2.0f * x);
  return 1.0f - 2.0f / (e + 1.0f);            // -> 1.0 cleanly as e -> inf
}

// ISA 7.12.2, 16-bit A 16x32: lane half hh: elems 0..7 = K[8*hh .. +8),
// elems 8..15 = K[16+8*hh .. +8). Row-major LDS tile, stride SRH.
__device__ __forceinline__ v16h load_frag_a(const _Float16* base, int row, int kofs, int hh) {
  F16x16 f;
  const _Float16* p = base + row * SRH + kofs + 8 * hh;
  f.h[0] = *(const v8h*)(p);
  f.h[1] = *(const v8h*)(p + 16);
  return f.v;
}

// 16-bit B 32x16: lane holds column n = weight row n; elems 0..7 = K[32*ks + 8*hh .. +8),
// elems 8..15 = K[32*ks + 16 + 8*hh .. +8) in the row-major W[out][in] matrix.
__device__ __forceinline__ v16h load_frag_b(const _Float16* wmat, int wrow, int ks, int hh) {
  F16x16 f;
  const _Float16* p = wmat + wrow * SRH + ks * 32 + 8 * hh;
  f.h[0] = *(const v8h*)(p);
  f.h[1] = *(const v8h*)(p + 16);
  return f.v;
}

__device__ __forceinline__ v8f wmma_f16(v16h a, v16h b, v8f c) {
  return __builtin_amdgcn_wmma_f32_16x16x32_f16(false, a, false, b, (short)0, c, false, false);
}

__launch_bounds__(NTHREADS, 1)
__global__ void gru_fused_kernel(GruParams p) {
  extern __shared__ char smem[];
  _Float16* w16  = (_Float16*)(smem + OFF_W);
  _Float16* x16  = (_Float16*)(smem + OFF_X);
  _Float16* h16  = (_Float16*)(smem + OFF_H16);
  _Float16* rh16 = (_Float16*)(smem + OFF_RH);

  const int tid  = threadIdx.x;
  const int wv   = tid >> 5;
  const int lane = tid & 31;
  const int hh   = lane >> 4;          // lane half (WMMA layout selector)
  const int l15  = lane & 15;
  const int n    = wv * 16 + l15;      // gate column owned by this lane in C/D
  const int bm0  = blockIdx.x * BM;
  const int T    = p.T;

  // ---- one-time staging: weights f32->f16, H0 -----------------------------
  {
    const float* wsrc[7] = {p.Wxz, p.Whz, p.Wxr, p.Whr, p.Wxh, p.Whh, p.Whq};
    for (int g = 0; g < 7; ++g) {
      const float* s = wsrc[g];
      _Float16* d = w16 + g * HS * SRH;
      for (int i = tid; i < HS * HS; i += NTHREADS) {
        int r = i >> 7, c = i & (HS - 1);
        d[r * SRH + c] = (_Float16)s[i];
      }
    }
    for (int i = tid; i < BM * HS; i += NTHREADS) {
      int r = i >> 7, c = i & (HS - 1);
      h16[r * SRH + c] = (_Float16)p.H0[(size_t)(bm0 + r) * HS + c];
    }
  }
  // H(f32) loop-carried in registers: lane owns H[m][n], m = v + 8*hh.
  float hreg[8];
#pragma unroll
  for (int v = 0; v < 8; ++v)
    hreg[v] = p.H0[(size_t)(bm0 + v + 8 * hh) * HS + n];

  // per-lane scalar biases (lane owns one gate column for all 8 C rows)
  const float bz = p.bxz[n] + p.bhz[n];
  const float br = p.bxr[n] + p.bhr[n];
  const float bh = p.bxh[n] + p.bhh[n];
  const float bq = p.bhq[n];

  const _Float16* Wz_x = w16 + 0 * HS * SRH;
  const _Float16* Wz_h = w16 + 1 * HS * SRH;
  const _Float16* Wr_x = w16 + 2 * HS * SRH;
  const _Float16* Wr_h = w16 + 3 * HS * SRH;
  const _Float16* Wh_x = w16 + 4 * HS * SRH;
  const _Float16* Wh_h = w16 + 5 * HS * SRH;
  const _Float16* Wq_h = w16 + 6 * HS * SRH;

  __syncthreads();

  // Loop-invariant output-projection B fragments: keep in registers forever.
  v16h bQ[KSTEPS];
#pragma unroll
  for (int ks = 0; ks < KSTEPS; ++ks) bQ[ks] = load_frag_b(Wq_h, n, ks, hh);

  // loop-carried A fragments of H (refreshed after each H update)
  v16h aH[KSTEPS];
#pragma unroll
  for (int ks = 0; ks < KSTEPS; ++ks) aH[ks] = load_frag_a(h16, l15, ks * 32, hh);

  // ---- X prefetch pipeline: this thread stages 8 consecutive f32 ----------
  const int xr = (tid * 8) >> 7;              // row in tile
  const int xc = (tid * 8) & (HS - 1);        // col in tile
  const float* xrow = p.X + (size_t)(bm0 + xr) * T * HS + xc;
  v4f px0 = *(const v4f*)(xrow);              // tile t = 0
  v4f px1 = *(const v4f*)(xrow + 4);

  for (int t = 0; t < T; ++t) {
    // ---- commit prefetched X tile to LDS (f32 -> f16) ---------------------
    {
      v8h hx;
      hx[0] = (_Float16)px0[0]; hx[1] = (_Float16)px0[1];
      hx[2] = (_Float16)px0[2]; hx[3] = (_Float16)px0[3];
      hx[4] = (_Float16)px1[0]; hx[5] = (_Float16)px1[1];
      hx[6] = (_Float16)px1[2]; hx[7] = (_Float16)px1[3];
      *(v8h*)(x16 + xr * SRH + xc) = hx;
      // issue global loads for t+1 now; latency hides under this whole step
      int t2 = (t + 1 < T) ? (t + 1) : t;
      const float* s = xrow + (size_t)t2 * HS;
      px0 = *(const v4f*)(s);
      px1 = *(const v4f*)(s + 4);
    }
    __syncthreads();                              // (A) x16 ready

    v16h aX[KSTEPS];
#pragma unroll
    for (int ks = 0; ks < KSTEPS; ++ks) aX[ks] = load_frag_a(x16, l15, ks * 32, hh);

    // ---- z / r gates: 4 independent WMMA chains of depth 4 ----------------
    v8f accZx = {}, accZh = {}, accRx = {}, accRh = {};
#pragma unroll
    for (int ks = 0; ks < KSTEPS; ++ks) {
      accZx = wmma_f16(aX[ks], load_frag_b(Wz_x, n, ks, hh), accZx);
      accZh = wmma_f16(aH[ks], load_frag_b(Wz_h, n, ks, hh), accZh);
      accRx = wmma_f16(aX[ks], load_frag_b(Wr_x, n, ks, hh), accRx);
      accRh = wmma_f16(aH[ks], load_frag_b(Wr_h, n, ks, hh), accRh);
    }
    float zf[8];
#pragma unroll
    for (int v = 0; v < 8; ++v) {
      zf[v]    = sigm(accZx[v] + accZh[v] + bz);
      float rf = sigm(accRx[v] + accRh[v] + br);
      int m = v + 8 * hh;                         // C layout: VGPR v -> row m
      rh16[m * SRH + n] = (_Float16)(rf * hreg[v]);
    }
    __syncthreads();                              // (B) rh16 ready

    // ---- candidate gate: 2 independent chains of depth 4 ------------------
    v8f accHx = {}, accHh = {};
#pragma unroll
    for (int ks = 0; ks < KSTEPS; ++ks) {
      v16h aRH = load_frag_a(rh16, l15, ks * 32, hh);
      accHx = wmma_f16(aX[ks], load_frag_b(Wh_x, n, ks, hh), accHx);
      accHh = wmma_f16(aRH,    load_frag_b(Wh_h, n, ks, hh), accHh);
    }
#pragma unroll
    for (int v = 0; v < 8; ++v) {
      float ht = tanh_fast(accHx[v] + accHh[v] + bh);
      float hn = zf[v] * hreg[v] + (1.0f - zf[v]) * ht;
      hreg[v] = hn;
      int m = v + 8 * hh;
      h16[m * SRH + n] = (_Float16)hn;            // f16 mirror for A operand
    }
    __syncthreads();                              // (D) new H visible

    // ---- output projection on H_{t+1}; reload doubles as next-step aH -----
#pragma unroll
    for (int ks = 0; ks < KSTEPS; ++ks) aH[ks] = load_frag_a(h16, l15, ks * 32, hh);
    v8f accQ = {};
#pragma unroll
    for (int ks = 0; ks < KSTEPS; ++ks)
      accQ = wmma_f16(aH[ks], bQ[ks], accQ);
#pragma unroll
    for (int v = 0; v < 8; ++v) {
      int m = v + 8 * hh;
      p.Y[((size_t)(bm0 + m) * T + t) * HS + n] = accQ[v] + bq;
    }
  }
}

extern "C" void kernel_launch(void* const* d_in, const int* in_sizes, int n_in,
                              void* d_out, int out_size, void* d_ws, size_t ws_size,
                              hipStream_t stream) {
  (void)n_in; (void)out_size; (void)d_ws; (void)ws_size;
  GruParams p;
  p.X   = (const float*)d_in[0];  p.H0  = (const float*)d_in[1];
  p.Wxz = (const float*)d_in[2];  p.bxz = (const float*)d_in[3];
  p.Whz = (const float*)d_in[4];  p.bhz = (const float*)d_in[5];
  p.Wxr = (const float*)d_in[6];  p.bxr = (const float*)d_in[7];
  p.Whr = (const float*)d_in[8];  p.bhr = (const float*)d_in[9];
  p.Wxh = (const float*)d_in[10]; p.bxh = (const float*)d_in[11];
  p.Whh = (const float*)d_in[12]; p.bhh = (const float*)d_in[13];
  p.Whq = (const float*)d_in[14]; p.bhq = (const float*)d_in[15];
  p.Y   = (float*)d_out;
  p.Bsz = in_sizes[1] / HS;                 // 512
  p.T   = in_sizes[0] / in_sizes[1];        // 1024

  hipFuncSetAttribute((const void*)gru_fused_kernel,
                      hipFuncAttributeMaxDynamicSharedMemorySize, SMEM_BYTES);
  gru_fused_kernel<<<p.Bsz / BM, NTHREADS, SMEM_BYTES, stream>>>(p);
}
